// LRP_PURE_layer_alldegree_54374285967907
// MI455X (gfx1250) — matrix operand order repaired and checked by reading, not compiled, not run on hardware
//
#include <hip/hip_runtime.h>
#include <hip/hip_fp16.h>

// ---------------- problem constants (match reference) ----------------
#define DD      64
#define LL      16
#define NPERM_C 250000
#define P_C     4000000          // NPERM * L
#define G_C     4096
#define TILES_C 15625            // NPERM / 16 (exact)

typedef __attribute__((ext_vector_type(16))) _Float16 v16h;
typedef __attribute__((ext_vector_type(8)))  _Float16 v8h;
typedef __attribute__((ext_vector_type(8)))  float    v8f;

#define WMMA(acc, a, b) \
  acc = __builtin_amdgcn_wmma_f32_16x16x32_f16(false, (a), false, (b), (short)0, (acc), false, false)

// ---------------- Tensor Data Mover staging (global -> LDS) -----------------
#if defined(__gfx1250__) && __has_builtin(__builtin_amdgcn_tensor_load_to_lds)
#define HAVE_TDM 1
typedef unsigned int u32x4 __attribute__((ext_vector_type(4)));
typedef int          i32x4 __attribute__((ext_vector_type(4)));
typedef int          i32x8 __attribute__((ext_vector_type(8)));

// 1-row 2D descriptor: ndw dwords, contiguous.  See CDNA5 ISA D# layout.
__device__ __forceinline__ void tdm_load(const void* gsrc, void* ldst, unsigned bytes) {
  unsigned long long ga = (unsigned long long)gsrc;
  unsigned lds = (unsigned)(unsigned long long)ldst;   // low 32 bits = LDS offset
  unsigned ndw = bytes >> 2;                           // data_size = 4B units
  u32x4 g0;
  g0[0] = 1u;                                          // count=1, user mode
  g0[1] = lds;                                         // lds_addr
  g0[2] = (unsigned)ga;                                // global_addr[31:0]
  g0[3] = (unsigned)((ga >> 32) & 0x01FFFFFFu) | (2u << 30);  // addr[56:32], type=2
  i32x8 g1;
  g1[0] = (int)(2u << 16);                             // wg_mask=0, data_size=2 (4B)
  g1[1] = (int)((ndw & 0xFFFFu) << 16);                // tensor_dim0[15:0] @ bits 63:48
  g1[2] = (int)((ndw >> 16) | (1u << 16));             // dim0[31:16], tensor_dim1=1 lo
  g1[3] = (int)((ndw & 0xFFFFu) << 16);                // dim1 hi=0, tile_dim0=ndw
  g1[4] = 1;                                           // tile_dim1=1, tile_dim2=0
  g1[5] = (int)ndw;                                    // tensor_dim0_stride[31:0]
  g1[6] = 0;                                           // stride hi, dim1_stride lo
  g1[7] = 0;
  i32x4 z4 = {0, 0, 0, 0};
#if __clang_major__ >= 23
  i32x8 z8 = {0, 0, 0, 0, 0, 0, 0, 0};
  __builtin_amdgcn_tensor_load_to_lds(g0, g1, z4, z4, z8, 0);
#else
  __builtin_amdgcn_tensor_load_to_lds(g0, g1, z4, z4, 0);
#endif
}
#endif

// Concatenate two contiguous 8-half runs into a 16-half A-fragment slice.
__device__ __forceinline__ v16h load_a16h(const _Float16* __restrict__ p0,
                                          const _Float16* __restrict__ p1) {
  v8h lo = *(const v8h*)(p0);
  v8h hi = *(const v8h*)(p1);
  return __builtin_shufflevector(lo, hi, 0, 1, 2, 3, 4, 5, 6, 7,
                                         8, 9, 10, 11, 12, 13, 14, 15);
}

// ---------------- zero ----------------
__global__ void zero_kernel(float4* __restrict__ p, long n4) {
  long i = (long)blockIdx.x * blockDim.x + threadIdx.x;
  long stride = (long)gridDim.x * blockDim.x;
  float4 z; z.x = z.y = z.z = z.w = 0.f;
  for (; i < n4; i += stride) p[i] = z;
}

// ---------------- weight repack into WMMA B-fragment layout (f16) ----------
// Wbig[k, c] = weights[b, c, l], k = l*64 + b.  Fragment layout:
// frag[kk][nt][lane][slot]: k = kk*32 + (slot<8 ? 8*(lane>>4)+slot : 16+8*(lane>>4)+slot-8), c = nt*16 + (lane&15)
__global__ void repack_wbig(const float* __restrict__ w, unsigned short* __restrict__ out) {
  int e = blockIdx.x * 256 + threadIdx.x;          // 65536 elements
  int slot = e & 15, lane = (e >> 4) & 31, nt = (e >> 9) & 3, kk = e >> 11;
  int k = kk * 32 + ((slot < 8) ? 8 * (lane >> 4) + slot : 16 + 8 * (lane >> 4) + slot - 8);
  int n = nt * 16 + (lane & 15);
  int l = k >> 6, b = k & 63;
  float v = w[b * (DD * LL) + n * LL + l];
  out[e] = __builtin_bit_cast(unsigned short, (_Float16)v);
}

// B2[k=c][n=c2] = Wl[c2, c] -> Wl[n*64 + k]
__global__ void repack_wl(const float* __restrict__ wl, unsigned short* __restrict__ out) {
  int e = blockIdx.x * 256 + threadIdx.x;          // 4096 elements
  int slot = e & 15, lane = (e >> 4) & 31, nt = (e >> 9) & 3, kk = e >> 11;
  int k = kk * 32 + ((slot < 8) ? 8 * (lane >> 4) + slot : 16 + 8 * (lane >> 4) + slot - 8);
  int n = nt * 16 + (lane & 15);
  out[e] = __builtin_bit_cast(unsigned short, (_Float16)wl[n * DD + k]);
}

// ---------------- scatter-add of gathered feature rows (packed f16 atomics) --
// 8 threads per nonzero, 4 half2 (8 channels) each -> global_atomic_pk_add_f16
__global__ void scatter_feat(const int* __restrict__ row, const int* __restrict__ col,
                             const float* __restrict__ val, const float* __restrict__ src,
                             __half2* __restrict__ dst, int nnz) {
  long t = (long)blockIdx.x * blockDim.x + threadIdx.x;
  int e = (int)(t >> 3);
  if (e >= nnz) return;
  int c0 = ((int)t & 7) * 8;
  int r = row[e]; float v = val[e];
  const float* s = src + (size_t)col[e] * DD + c0;
  __half2* d = dst + ((size_t)r * DD + c0) / 2;
  #pragma unroll
  for (int i = 0; i < 4; ++i) {
    __half2 h = __floats2half2_rn(v * s[2 * i], v * s[2 * i + 1]);
    unsafeAtomicAdd(d + i, h);
  }
}

// degree scatter: only slots p%16 in {0,5,10,15} are ever read
__global__ void scatter_deg(const int* __restrict__ row, const int* __restrict__ col,
                            const float* __restrict__ val, const float* __restrict__ degs,
                            float* __restrict__ degws, int nnz) {
  int e = blockIdx.x * blockDim.x + threadIdx.x;
  if (e >= nnz) return;
  int r = row[e];
  if (((r & 15) % 5) != 0) return;                 // keep l = 0,5,10,15
  unsafeAtomicAdd(&degws[r], val[e] * degs[col[e]]);
}

// ---------------- degree MLP gate: dg = relu(deg4 @ W0.T + b0) @ W1.T + b1 ----
__global__ __launch_bounds__(256) void dg_kernel(const float* __restrict__ degws,
                                                 const float* __restrict__ W0,
                                                 const float* __restrict__ b0,
                                                 const float* __restrict__ W1,
                                                 const float* __restrict__ b1,
                                                 float* __restrict__ dgout) {
  __shared__ float t[8][128];
  int wave = threadIdx.x >> 5, lane = threadIdx.x & 31;
  int n = blockIdx.x * 8 + wave;
  if (n >= NPERM_C) n = NPERM_C - 1;               // duplicate write, benign
  const float* dp = degws + (size_t)n * 16;
  float d0 = dp[0], d1 = dp[5], d2 = dp[10], d3 = dp[15];
  #pragma unroll
  for (int jj = 0; jj < 4; ++jj) {
    int j = lane * 4 + jj;
    float v = b0[j] + W0[j*4+0]*d0 + W0[j*4+1]*d1 + W0[j*4+2]*d2 + W0[j*4+3]*d3;
    t[wave][j] = v > 0.f ? v : 0.f;
  }
  __syncthreads();
  #pragma unroll
  for (int cc = 0; cc < 2; ++cc) {
    int c = lane + cc * 32;
    float acc = b1[c];
    #pragma unroll 8
    for (int j = 0; j < 128; ++j) acc += t[wave][j] * W1[c * 128 + j];
    dgout[(size_t)n * DD + c] = acc;
  }
}

// ---------------- fused WMMA GEMM: h=A@Wbig; h2=relu(h)@Wl.T; out=(h2+bl)*dg ----
__global__ __launch_bounds__(256) void gemm_fused(const _Float16* __restrict__ nfeat,
                                                  const unsigned short* __restrict__ wbigf,
                                                  const unsigned short* __restrict__ wlf,
                                                  const float* __restrict__ bias,
                                                  const float* __restrict__ bl,
                                                  const float* __restrict__ dg,
                                                  float* __restrict__ outbuf) {
  extern __shared__ char smem[];
  unsigned short* ldsB  = (unsigned short*)smem;            // 128 KB (phase 1)
  unsigned short* ldsWl = (unsigned short*)(smem + 131072); // 8 KB
  _Float16*       ldsH  = (_Float16*)smem;                  // phase 2 reuse (16 KB)

  const int tid = threadIdx.x;
#if defined(HAVE_TDM)
  if (tid == 0) {                         // TDM DMA: weights -> LDS, async
    tdm_load(wbigf, ldsB, 131072u);
    tdm_load(wlf, ldsWl, 8192u);
    __builtin_amdgcn_s_wait_tensorcnt(0);
  }
#else
  {
    const uint4* s0 = (const uint4*)wbigf; uint4* d0 = (uint4*)ldsB;
    #pragma unroll 4
    for (int i = tid; i < 8192; i += 256) d0[i] = s0[i];
    const uint4* s1 = (const uint4*)wlf;   uint4* d1 = (uint4*)ldsWl;
    for (int i = tid; i < 512; i += 256) d1[i] = s1[i];
  }
#endif
  __syncthreads();

  const int wave = tid >> 5, lane = tid & 31;
  int tile = blockIdx.x * 8 + wave;
  if (tile > TILES_C - 1) tile = TILES_C - 1;      // duplicate identical writes: benign
  const int m   = lane & 15;
  const int klo = (lane < 16) ? 0 : 8;
  const int rowbase = tile * 16;
  const _Float16* arow = nfeat + (size_t)(rowbase + m) * 1024;
  const v16h* Bf = (const v16h*)ldsB;

  v8f c0 = {}, c1 = {}, c2 = {}, c3 = {};
  #pragma unroll 4
  for (int kk = 0; kk < 32; ++kk) {
    const int kb = kk * 32 + klo;
    if ((kk & 7) == 0) __builtin_prefetch(arow + kb + 512, 0, 0);  // global_prefetch_b8
    v16h a = load_a16h(arow + kb, arow + kb + 16);
    const v16h* bp = Bf + kk * 128 + lane;
    v16h b0 = bp[0], b1 = bp[32], b2 = bp[64], b3 = bp[96];
    WMMA(c0, a, b0); WMMA(c1, a, b1); WMMA(c2, a, b2); WMMA(c3, a, b3);
  }
  __syncthreads();   // every wave done reading ldsB before reuse as H

  // relu(h + bias) -> LDS f16 (C layout: elem r => row rb+r, col nt*16+nc)
  _Float16* hws = ldsH + wave * 1024;
  const int rb = (lane < 16) ? 0 : 8;
  const int nc = lane & 15;
  #define EPI1(acc, nt) { \
      const int col = (nt)*16 + nc; const float bv = bias[col]; \
      _Pragma("unroll") for (int r = 0; r < 8; ++r) { \
        float v = (acc)[r] + bv; hws[(rb + r) * 64 + col] = (_Float16)(v > 0.f ? v : 0.f); } }
  EPI1(c0, 0) EPI1(c1, 1) EPI1(c2, 2) EPI1(c3, 3)
  __syncthreads();

  // phase 2: h2 = relu_h[16x64] @ Wl^T  (K = 64 -> 2 chunks)
  const v16h* Wf = (const v16h*)ldsWl;
  const _Float16* hrow = hws + m * 64;
  v8f e0 = {}, e1 = {}, e2 = {}, e3 = {};
  #pragma unroll
  for (int kk = 0; kk < 2; ++kk) {
    const int kb = kk * 32 + klo;
    v16h a = load_a16h(hrow + kb, hrow + kb + 16);
    const v16h* wp = Wf + kk * 128 + lane;
    v16h b0 = wp[0], b1 = wp[32], b2 = wp[64], b3 = wp[96];
    WMMA(e0, a, b0); WMMA(e1, a, b1); WMMA(e2, a, b2); WMMA(e3, a, b3);
  }

  // epilogue: (+bl) * dg -> outbuf
  #define EPI2(acc, nt) { \
      const int col = (nt)*16 + nc; const float bv = bl[col]; \
      _Pragma("unroll") for (int r = 0; r < 8; ++r) { \
        const size_t g = (size_t)(rowbase + rb + r) * 64 + col; \
        outbuf[g] = ((acc)[r] + bv) * dg[g]; } }
  EPI2(e0, 0) EPI2(e1, 1) EPI2(e2, 2) EPI2(e3, 3)
}

// ---------------- pooling scatter: out[pool_row[i]] += outbuf[pool_col[i]] -----
__global__ void pool_kernel(const int* __restrict__ prow, const int* __restrict__ pcol,
                            const float* __restrict__ pval, const float* __restrict__ src,
                            float* __restrict__ out, int nnz) {
  long t = (long)blockIdx.x * blockDim.x + threadIdx.x;
  int e = (int)(t >> 4);
  if (e >= nnz) return;
  int c0 = ((int)t & 15) * 4;
  float v = pval[e];
  const float* s = src + (size_t)pcol[e] * DD + c0;
  float* d = out + (size_t)prow[e] * DD + c0;
  unsafeAtomicAdd(d + 0, v * s[0]);
  unsafeAtomicAdd(d + 1, v * s[1]);
  unsafeAtomicAdd(d + 2, v * s[2]);
  unsafeAtomicAdd(d + 3, v * s[3]);
}

extern "C" void kernel_launch(void* const* d_in, const int* in_sizes, int n_in,
                              void* d_out, int out_size, void* d_ws, size_t ws_size,
                              hipStream_t stream) {
  const float* x      = (const float*)d_in[0];
  const float* efeat  = (const float*)d_in[1];
  const float* degs   = (const float*)d_in[2];
  const int*   n2pr   = (const int*)d_in[3];
  const int*   n2pc   = (const int*)d_in[4];
  const float* n2pv   = (const float*)d_in[5];
  const int*   e2pr   = (const int*)d_in[6];
  const int*   e2pc   = (const int*)d_in[7];
  const float* e2pv   = (const float*)d_in[8];
  const int*   poolr  = (const int*)d_in[9];
  const int*   poolc  = (const int*)d_in[10];
  const float* poolv  = (const float*)d_in[11];
  const float* weights= (const float*)d_in[12];
  const float* bias   = (const float*)d_in[13];
  const float* W0     = (const float*)d_in[14];
  const float* b0     = (const float*)d_in[15];
  const float* W1     = (const float*)d_in[16];
  const float* b1     = (const float*)d_in[17];
  const float* Wl     = (const float*)d_in[18];
  const float* bl     = (const float*)d_in[19];
  float* out = (float*)d_out;

  char* ws = (char*)d_ws;
  __half2* nfeat_h2      = (__half2*)(ws);                    // P*64*2   = 512,000,000 B
  _Float16* nfeat_h      = (_Float16*)(ws);
  float* degws           = (float*)(ws + 512000000ll);        // P*4      =  16,000,000 B
  float* outbuf          = (float*)(ws + 528000000ll);        // NPERM*64*4 = 64,000,000 B
  float* dgbuf           = (float*)(ws + 592000000ll);        // NPERM*64*4 = 64,000,000 B
  unsigned short* wbigf  = (unsigned short*)(ws + 656000000ll); // 131,072 B
  unsigned short* wlf    = (unsigned short*)(ws + 656131072ll); //   8,192 B

  // zero nfeat(f16) + degws (contiguous 528 MB) and d_out
  zero_kernel<<<4096, 256, 0, stream>>>((float4*)ws, 33000000l);
  zero_kernel<<<64, 256, 0, stream>>>((float4*)out, (long)(G_C * DD / 4));

  // weight repack to f16 fragments
  repack_wbig<<<256, 256, 0, stream>>>(weights, wbigf);
  repack_wl<<<16, 256, 0, stream>>>(Wl, wlf);

  // scatter node/edge features into perm slots (packed f16 atomics)
  scatter_feat<<<P_C / 32, 256, 0, stream>>>(n2pr, n2pc, n2pv, x, nfeat_h2, P_C);
  scatter_feat<<<P_C / 32, 256, 0, stream>>>(e2pr, e2pc, e2pv, efeat, nfeat_h2, P_C);
  scatter_deg<<<P_C / 256, 256, 0, stream>>>(n2pr, n2pc, n2pv, degs, degws, P_C);

  // degree gate
  dg_kernel<<<NPERM_C / 8, 256, 0, stream>>>(degws, W0, b0, W1, b1, dgbuf);

  // fused WMMA GEMM chain (dynamic LDS: 128KB Wbig/H-reuse + 8KB Wl)
  gemm_fused<<<(TILES_C + 7) / 8, 256, 139264, stream>>>(nfeat_h, wbigf, wlf, bias, bl,
                                                         dgbuf, outbuf);

  // pooling scatter into [G, 64]
  pool_kernel<<<NPERM_C / 16, 256, 0, stream>>>(poolr, poolc, poolv, outbuf, out, NPERM_C);
}
